// CustomModel_67671504716239
// MI455X (gfx1250) — matrix operand order repaired and checked by reading, not compiled, not run on hardware
//
#include <hip/hip_runtime.h>

typedef __attribute__((ext_vector_type(16))) _Float16 v16h;
typedef __attribute__((ext_vector_type(8)))  _Float16 v8h;
typedef __attribute__((ext_vector_type(8)))  float    v8f;

#define RESXF   2048.0f
#define DIM1    2048
#define DIM2    512
#define FEAT    32
#define HWPTS   (1024 * 1024)
#define NTILES  (HWPTS / 16)

// DPP16 control codes
#define DPP_QUAD_XOR1   0xB1   // quad_perm [1,0,3,2]  : lane ^= 1
#define DPP_QUAD_XOR3   0x1B   // quad_perm [3,2,1,0]  : lane ^= 3
#define DPP_HALF_MIRROR 0x141  // lane -> lane ^ 7 within 8
#define DPP_ROW_MIRROR  0x140  // lane -> lane ^ 15 within 16

static __device__ __forceinline__ int clampi(int v, int lo, int hi) {
    return v < lo ? lo : (v > hi ? hi : v);
}

static __device__ __forceinline__ v8h splat8(_Float16 x) {
    v8h v;
#pragma unroll
    for (int i = 0; i < 8; ++i) v[i] = x;
    return v;
}

// x + dpp_permute(x): VALU-only cross-lane add
template <int CTRL>
static __device__ __forceinline__ float dpp_add(float x) {
    int p = __builtin_amdgcn_update_dpp(0, __float_as_int(x), CTRL, 0xf, 0xf, true);
    return x + __int_as_float(p);
}

// Butterfly merge step: lanes with `bit` clear end up accumulating `a`'s row,
// lanes with `bit` set accumulate `b`'s row. The DPP permutation must flip the
// selection bit; flipping already-consumed lower bits is harmless since the
// row choice is evaluated per-lane before the move.
template <int CTRL>
static __device__ __forceinline__ float dpp_merge(float a, float b, bool bit) {
    float self  = bit ? b : a;
    float other = bit ? a : b;
    int p = __builtin_amdgcn_update_dpp(0, __float_as_int(other), CTRL, 0xf, 0xf, true);
    return self + __int_as_float(p);
}

// ---------------------------------------------------------------------------
// Transpose [F=32, N] f32 (feature-major) -> [N, 32] f16 (point-major).
// ---------------------------------------------------------------------------
__global__ void transpose_to_f16(const float* __restrict__ src,
                                 _Float16* __restrict__ dst, int n) {
    int p = blockIdx.x * blockDim.x + threadIdx.x;
    if (p >= n) return;
    union { _Float16 h[32]; uint4 u[4]; } buf;
#pragma unroll
    for (int f = 0; f < 32; ++f) buf.h[f] = (_Float16)src[f * n + p];
    uint4* out = (uint4*)(dst + (size_t)p * 32);
    out[0] = buf.u[0]; out[1] = buf.u[1]; out[2] = buf.u[2]; out[3] = buf.u[3];
}

// ---------------------------------------------------------------------------
// Fused sample + MLP. One wave32 handles 16 query points per tile.
// ---------------------------------------------------------------------------
__global__ __launch_bounds__(256) void fused_sample_mlp(
    const float* __restrict__ coords,
    const float* __restrict__ W1, const float* __restrict__ b1,
    const float* __restrict__ W2, const float* __restrict__ b2,
    const _Float16* __restrict__ tlx,   // [2048][32] f16
    const _Float16* __restrict__ tly,   // [2048][32] f16
    const _Float16* __restrict__ tpl,   // [512*512][32] f16
    float* __restrict__ out)
{
    const int lane  = threadIdx.x & 31;
    const int n16   = lane & 15;
    const int hi    = lane >> 4;        // 0: lanes 0-15, 1: lanes 16-31
    const int kbase = hi ? 8 : 0;       // B-fragment K offset for this half
    const int fb    = hi ? 8 : 0;       // A-fragment feature-group base
    const bool b1l  = (lane & 2) != 0;  // selection bits for merge tree
    const bool b2l  = (lane & 4) != 0;
    const bool b3l  = (lane & 8) != 0;
    const bool evn  = (lane & 1) == 0;
    const int  oidx = hi * 8 + (n16 >> 1);  // output row this lane ends up with

    // ---- Pack W1/W2 into B fragments + bias C fragments (loaded once) ----
    // The 1/64 mean factor is folded into W1/b1 here.
    v16h bw1[4], bw2[4];
    v8f  cb1[4], cb2[4];
#pragma unroll
    for (int c = 0; c < 4; ++c) {
        const int m = c * 16 + n16;
        const float* r1 = W1 + m * FEAT + kbase;
        const float* r2 = W2 + m * FEAT + kbase;
#pragma unroll
        for (int i = 0; i < 16; ++i) {
            bw1[c][i] = (_Float16)(r1[i + (i & 8)] * (1.0f / 64.0f));
            bw2[c][i] = (_Float16)r2[i + (i & 8)];
        }
        const float bb1 = b1[m] * (1.0f / 64.0f), bb2 = b2[m];
#pragma unroll
        for (int i = 0; i < 8; ++i) { cb1[c][i] = bb1; cb2[c][i] = bb2; }
    }

    const int nwaves = (gridDim.x * blockDim.x) >> 5;
    const int wid    = (blockIdx.x * blockDim.x + threadIdx.x) >> 5;

    for (int t = wid; t < NTILES; t += nwaves) {
        const int p  = t * 16 + n16;
        const float cx = coords[2 * p + 0];
        const float cy = coords[2 * p + 1];
        const float gx = cx * (2.0f / RESXF) - 1.0f;
        const float gy = cy * (2.0f / RESXF) - 1.0f;

        // ---- 1-D line samples (border padding, align_corners=True) ----
        float posx = fminf(fmaxf((gx + 1.0f) * 0.5f * (float)(DIM1 - 1), 0.0f),
                           (float)(DIM1 - 1));
        float posy = fminf(fmaxf((gy + 1.0f) * 0.5f * (float)(DIM1 - 1), 0.0f),
                           (float)(DIM1 - 1));
        int ix = clampi((int)floorf(posx), 0, DIM1 - 2);
        int iy = clampi((int)floorf(posy), 0, DIM1 - 2);
        v8h frxv = splat8((_Float16)(posx - (float)ix));
        v8h fryv = splat8((_Float16)(posy - (float)iy));

        const _Float16* bx0 = tlx + ix * FEAT + fb;
        const _Float16* by0 = tly + iy * FEAT + fb;
        v8h x0a = *(const v8h*)(bx0);
        v8h x0b = *(const v8h*)(bx0 + 16);
        v8h x1a = *(const v8h*)(bx0 + FEAT);
        v8h x1b = *(const v8h*)(bx0 + FEAT + 16);
        v8h y0a = *(const v8h*)(by0);
        v8h y0b = *(const v8h*)(by0 + 16);
        v8h y1a = *(const v8h*)(by0 + FEAT);
        v8h y1b = *(const v8h*)(by0 + FEAT + 16);

        v8h lxa = x0a + (x1a - x0a) * frxv;   // feat_x, group A (feats fb..fb+7)
        v8h lxb = x0b + (x1b - x0b) * frxv;   // feat_x, group B (feats fb+16..fb+23)
        v8h lya = y0a + (y1a - y0a) * fryv;
        v8h lyb = y0b + (y1b - y0b) * fryv;

        // ---- 2-D plane sample (coords in-range -> all 4 taps valid) ----
        float px = (gx + 1.0f) * 0.5f * (float)(DIM2 - 1);
        float py = (gy + 1.0f) * 0.5f * (float)(DIM2 - 1);
        int x0 = clampi((int)floorf(px), 0, DIM2 - 2);
        int y0 = clampi((int)floorf(py), 0, DIM2 - 2);
        float fx = px - (float)x0;
        float fy = py - (float)y0;
        v8h w00 = splat8((_Float16)((1.0f - fx) * (1.0f - fy)));
        v8h w01 = splat8((_Float16)(fx * (1.0f - fy)));
        v8h w10 = splat8((_Float16)((1.0f - fx) * fy));
        v8h w11 = splat8((_Float16)(fx * fy));

        const _Float16* bp = tpl + (y0 * DIM2 + x0) * FEAT + fb;
        v8h p00a = *(const v8h*)(bp);
        v8h p00b = *(const v8h*)(bp + 16);
        v8h p01a = *(const v8h*)(bp + FEAT);
        v8h p01b = *(const v8h*)(bp + FEAT + 16);
        v8h p10a = *(const v8h*)(bp + DIM2 * FEAT);
        v8h p10b = *(const v8h*)(bp + DIM2 * FEAT + 16);
        v8h p11a = *(const v8h*)(bp + DIM2 * FEAT + FEAT);
        v8h p11b = *(const v8h*)(bp + DIM2 * FEAT + FEAT + 16);

        v8h lpa = p00a * w00 + p01a * w01 + p10a * w10 + p11a * w11;
        v8h lpb = p00b * w00 + p01b * w01 + p10b * w10 + p11b * w11;

        // combined = feat_x * feat_y + feat_p  -> A fragment (register concat)
        v8h za = lxa * lya + lpa;
        v8h zb = lxb * lyb + lpb;
        v16h a = __builtin_shufflevector(za, zb,
                                         0, 1, 2, 3, 4, 5, 6, 7,
                                         8, 9, 10, 11, 12, 13, 14, 15);

        // ---- MLP: 4 neuron chunks x (h1, gate) via WMMA, masked accumulate ----
        float acc[8];
        {
            v8f h = __builtin_amdgcn_wmma_f32_16x16x32_f16(
                false, a, false, bw1[0], (short)0, cb1[0], false, false);
            v8f g = __builtin_amdgcn_wmma_f32_16x16x32_f16(
                false, a, false, bw2[0], (short)0, cb2[0], false, false);
#pragma unroll
            for (int r = 0; r < 8; ++r)
                acc[r] = (g[r] > 0.0f) ? h[r] : 0.0f;
        }
#pragma unroll
        for (int c = 1; c < 4; ++c) {
            v8f h = __builtin_amdgcn_wmma_f32_16x16x32_f16(
                false, a, false, bw1[c], (short)0, cb1[c], false, false);
            v8f g = __builtin_amdgcn_wmma_f32_16x16x32_f16(
                false, a, false, bw2[c], (short)0, cb2[c], false, false);
#pragma unroll
            for (int r = 0; r < 8; ++r)
                acc[r] += (g[r] > 0.0f) ? h[r] : 0.0f;
        }

        // ---- mean over 64 neurons: DPP merge-tree within each 16-lane half ----
        // level 1 (sel bit3, xor15): 8 -> 4
        float t0 = dpp_merge<DPP_ROW_MIRROR>(acc[0], acc[4], b3l);
        float t1 = dpp_merge<DPP_ROW_MIRROR>(acc[1], acc[5], b3l);
        float t2 = dpp_merge<DPP_ROW_MIRROR>(acc[2], acc[6], b3l);
        float t3 = dpp_merge<DPP_ROW_MIRROR>(acc[3], acc[7], b3l);
        // level 2 (sel bit2, xor7): 4 -> 2
        float u0 = dpp_merge<DPP_HALF_MIRROR>(t0, t2, b2l);
        float u1 = dpp_merge<DPP_HALF_MIRROR>(t1, t3, b2l);
        // level 3 (sel bit1, xor3): 2 -> 1
        float v = dpp_merge<DPP_QUAD_XOR3>(u0, u1, b1l);
        // level 4 (xor1): finish 16-lane sum
        v = dpp_add<DPP_QUAD_XOR1>(v);

        // lane (n16) holds row n16>>1 of its half; even lanes store one dword
        // -> the tile's whole 64B output line in a single coalesced request.
        if (evn) out[t * 16 + oidx] = v;
    }
}

// ---------------------------------------------------------------------------
// Workspace layout (f16): tlx [2048*32] | tly [2048*32] | tpl [512*512*32]
// Total ~16.25 MB.
// ---------------------------------------------------------------------------
extern "C" void kernel_launch(void* const* d_in, const int* in_sizes, int n_in,
                              void* d_out, int out_size, void* d_ws, size_t ws_size,
                              hipStream_t stream) {
    const float* coords = (const float*)d_in[0];
    const float* line_x = (const float*)d_in[1];
    const float* line_y = (const float*)d_in[2];
    const float* plane  = (const float*)d_in[3];
    const float* W1     = (const float*)d_in[4];
    const float* b1     = (const float*)d_in[5];
    const float* W2     = (const float*)d_in[6];
    const float* b2     = (const float*)d_in[7];
    float* out = (float*)d_out;

    _Float16* tlx = (_Float16*)d_ws;
    _Float16* tly = tlx + 2048 * 32;
    _Float16* tpl = tly + 2048 * 32;

    transpose_to_f16<<<(2048 + 255) / 256, 256, 0, stream>>>(line_x, tlx, 2048);
    transpose_to_f16<<<(2048 + 255) / 256, 256, 0, stream>>>(line_y, tly, 2048);
    transpose_to_f16<<<(512 * 512 + 255) / 256, 256, 0, stream>>>(plane, tpl, 512 * 512);

    fused_sample_mlp<<<1024, 256, 0, stream>>>(coords, W1, b1, W2, b2,
                                               tlx, tly, tpl, out);
}
